// MultiHeadAttentionBlock_84198538871405
// MI455X (gfx1250) — matrix operand order repaired and checked by reading, not compiled, not run on hardware
//
#include <hip/hip_runtime.h>
#include <stdint.h>

#define DEV __device__ __forceinline__

typedef unsigned short u16;
typedef __attribute__((ext_vector_type(16))) __bf16 v16bf;
typedef __attribute__((ext_vector_type(8)))  float  v8f;
typedef __attribute__((ext_vector_type(4)))  unsigned int u32x4;
typedef __attribute__((ext_vector_type(8)))  int i32x8;
typedef __attribute__((ext_vector_type(4)))  int i32x4;

struct alignas(16) Chunk  { unsigned u[4]; };
struct alignas(16) Chunk2 { Chunk a, b; };

DEV u16 f2bf(float f) {
    unsigned u = __builtin_bit_cast(unsigned, f);
    u += 0x7FFFu + ((u >> 16) & 1u);           // round-to-nearest-even
    return (u16)(u >> 16);
}

DEV v8f vzero() {
    v8f z = {0.f, 0.f, 0.f, 0.f, 0.f, 0.f, 0.f, 0.f};
    return z;
}

// Load a 32-byte WMMA fragment from two 16-byte chunks.
DEV v16bf ld_frag(const u16* p0, const u16* p1) {
    Chunk2 t;
    t.a = *(const Chunk*)p0;
    t.b = *(const Chunk*)p1;
    return __builtin_bit_cast(v16bf, t);
}

DEV v8f wmma_bf(v16bf a, v16bf b, v8f c) {
    return __builtin_amdgcn_wmma_f32_16x16x32_bf16(
        /*neg_a=*/false, a, /*neg_b=*/false, b,
        /*c_mod=*/(short)0, c, /*reuse_a=*/false, /*reuse_b=*/false);
}

// ---------------------------------------------------------------------------
// Tensor Data Mover: 2-D tile (bf16) Global -> LDS.  D# built per ISA 08 §8.
// Group0: count=1 | lds_addr | global_addr | type=2
// Group1: data_size=2B, tensor_dim0/1, tile_dim0/1, tensor_dim0_stride
// 6-arg builtin form (clang-23 / therock-10.0 toolchain).
// ---------------------------------------------------------------------------
DEV void tdm_load_2d(unsigned lds_addr, const void* gptr,
                     unsigned tile_k, unsigned tile_rows,
                     unsigned tdim0, unsigned tdim1, unsigned stride0)
{
    unsigned long long ga = (unsigned long long)(uintptr_t)gptr;
    u32x4 g0;
    g0[0] = 1u;                                            // count=1, user mode
    g0[1] = lds_addr;                                      // LDS byte address
    g0[2] = (unsigned)ga;                                  // global_addr[31:0]
    g0[3] = (unsigned)((ga >> 32) & 0x01FFFFFFull) | (2u << 30);  // addr[56:32] | type=2
    i32x8 g1;
    g1[0] = (int)(1u << 16);                               // wg_mask=0, data_size=1 (2 bytes)
    g1[1] = (int)((tdim0 & 0xFFFFu) << 16);                // tensor_dim0[15:0] @ bits 63:48
    g1[2] = (int)(((tdim0 >> 16) & 0xFFFFu) | ((tdim1 & 0xFFFFu) << 16));
    g1[3] = (int)(((tdim1 >> 16) & 0xFFFFu) | (tile_k << 16));    // tile_dim0
    g1[4] = (int)tile_rows;                                // tile_dim1 (tile_dim2=0)
    g1[5] = (int)stride0;                                  // tensor_dim0_stride[31:0]
    g1[6] = 0;
    g1[7] = 0;
    i32x4 gz4 = {0, 0, 0, 0};                              // groups 2/3 unused (2-D)
    i32x8 gz8 = {0, 0, 0, 0, 0, 0, 0, 0};
    __builtin_amdgcn_tensor_load_to_lds(g0, g1, gz4, gz4, gz8, 0);
}

// ---------------------------------------------------------------------------
// f32 -> bf16 conversion (4 elems / thread, b128 loads)
// ---------------------------------------------------------------------------
__global__ void cvt_bf16_kernel(const float* __restrict__ s, u16* __restrict__ d, int n4)
{
    int i = blockIdx.x * blockDim.x + threadIdx.x;
    if (i < n4) {
        float4 f = ((const float4*)s)[i];
        ushort4 o;
        o.x = f2bf(f.x); o.y = f2bf(f.y); o.z = f2bf(f.z); o.w = f2bf(f.w);
        ((ushort4*)d)[i] = o;
    }
}

// ---------------------------------------------------------------------------
// bf16 GEMM  C[M,N] = A[M,K] * W[N,K]^T (+bias), K=1024, block tile 64x256,
// 8 waves of 32x64 each.  A/B tiles double-buffered in LDS; DMA of stage k+1
// (TDM) overlaps WMMA compute of stage k; pipelined with s_wait_tensorcnt(2).
// MODE 0: fused QKV epilogue -> Q,K row-major [B,H,S,Dk] bf16, V transposed
//         [B,H,Dk,S] bf16, with per-section bias.
// MODE 1: f32 output [M,N] + bias.
// ---------------------------------------------------------------------------
template<int MODE>
__global__ void __launch_bounds__(256)
gemm_bf16(const u16* __restrict__ A, const u16* __restrict__ W,
          const float* __restrict__ bq, const float* __restrict__ bk,
          const float* __restrict__ bv,
          u16* __restrict__ Qo, u16* __restrict__ Ko, u16* __restrict__ Vto,
          float* __restrict__ Fo, int Ntot)
{
    __shared__ __align__(128) u16 As[2][64][32];     //  8 KB (double buffer)
    __shared__ __align__(128) u16 Bs[2][256][32];    // 32 KB (double buffer)

    const int lane = threadIdx.x & 31;
    const int wv   = threadIdx.x >> 5;
    const int wm   = wv >> 2;                     // 0..1  -> M offset 32*wm
    const int wn   = wv & 3;                      // 0..3  -> N offset 64*wn
    const int hi   = lane >> 4;
    const int ln   = lane & 15;

    const int m0 = blockIdx.y * 64;
    const int n0 = blockIdx.x * 256;

    v8f acc[2][4];
#pragma unroll
    for (int t = 0; t < 2; ++t)
#pragma unroll
        for (int j = 0; j < 4; ++j) acc[t][j] = vzero();

    unsigned asAddr[2], bsAddr[2];
#pragma unroll
    for (int p = 0; p < 2; ++p) {
        asAddr[p] = (unsigned)(uintptr_t)(const void*)&As[p][0][0];
        bsAddr[p] = (unsigned)(uintptr_t)(const void*)&Bs[p][0][0];
    }

    const u16* Ab = A + (size_t)m0 * 1024;
    const u16* Wb = W + (size_t)n0 * 1024;

    if (threadIdx.x < 32) {                       // prologue: stage 0 DMAs
        __builtin_prefetch(bq, 0, 0);
        tdm_load_2d(asAddr[0], Ab, 32,  64, 1024, 4096u,          1024);
        tdm_load_2d(bsAddr[0], Wb, 32, 256, 1024, (unsigned)Ntot, 1024);
    }

    for (int st = 0; st < 32; ++st) {
        if (threadIdx.x < 32) {                   // overlap: DMA stage st+1
            if (st + 1 < 32) {
                const int k1 = (st + 1) * 32;
                const int nb = (st + 1) & 1;
                tdm_load_2d(asAddr[nb], Ab + k1, 32,  64, 1024, 4096u,          1024);
                tdm_load_2d(bsAddr[nb], Wb + k1, 32, 256, 1024, (unsigned)Ntot, 1024);
                __builtin_amdgcn_s_wait_tensorcnt(2);   // stage st complete
            } else {
                __builtin_amdgcn_s_wait_tensorcnt(0);
            }
        }
        __syncthreads();                          // stage st visible to all waves

        const int cb = st & 1;
        v16bf af[2], bfr[4];
#pragma unroll
        for (int t = 0; t < 2; ++t) {             // A frag: rows, k {0..7,16..23}+8*hi
            const u16* r = &As[cb][wm * 32 + t * 16 + ln][hi * 8];
            af[t] = ld_frag(r, r + 16);
        }
#pragma unroll
        for (int j = 0; j < 4; ++j) {             // B frag: row n, k {0..7,16..23}+8*hi
            const u16* r = &Bs[cb][wn * 64 + j * 16 + ln][hi * 8];
            bfr[j] = ld_frag(r, r + 16);
        }
#pragma unroll
        for (int t = 0; t < 2; ++t)
#pragma unroll
            for (int j = 0; j < 4; ++j)
                acc[t][j] = wmma_bf(af[t], bfr[j], acc[t][j]);

        __syncthreads();                          // buffer reusable for st+2 DMA
    }

    // Epilogue.  C/D layout: element r -> (m = hi*8 + r, n = ln)
#pragma unroll
    for (int t = 0; t < 2; ++t) {
#pragma unroll
        for (int j = 0; j < 4; ++j) {
#pragma unroll
            for (int r = 0; r < 8; ++r) {
                int mg = m0 + wm * 32 + t * 16 + hi * 8 + r;
                int ng = n0 + wn * 64 + j * 16 + ln;
                float v = acc[t][j][r];
                if (MODE == 0) {
                    int sect = ng >> 10;           // 0:Q 1:K 2:V
                    int nn   = ng & 1023;
                    v += (sect == 0) ? bq[nn] : (sect == 1 ? bk[nn] : bv[nn]);
                    int b = mg >> 11, s = mg & 2047;
                    int hh = nn >> 6,  d = nn & 63;
                    size_t bh = (size_t)(b * 16 + hh);
                    u16 w = f2bf(v);
                    if (sect == 0)      Qo [(bh * 2048 + s) * 64 + d]   = w;
                    else if (sect == 1) Ko [(bh * 2048 + s) * 64 + d]   = w;
                    else                Vto[(bh * 64 + d) * 2048 + s]   = w;
                } else {
                    Fo[(size_t)mg * 1024 + ng] = v + bq[ng];
                }
            }
        }
    }
}

// ---------------------------------------------------------------------------
// Causal flash attention.  Block = 8 waves sharing one (b,h) and a 128-row
// q band; each wave owns a 16-row q tile.  K (32x64) and V^T (64x32) tiles
// staged once per block via double-buffered TDM; inner loop is LDS + WMMA.
// Q,K: [B,H,S,64] bf16 row-major.  Vt: [B,H,64,S] bf16.  O: [B*S,1024] bf16.
// ---------------------------------------------------------------------------
__global__ void __launch_bounds__(256)
attn_fwd(const u16* __restrict__ Q, const u16* __restrict__ K,
         const u16* __restrict__ Vt, u16* __restrict__ O)
{
    __shared__ __align__(128) u16 Ks[2][32][64];     // [key][d]   4 KB x2
    __shared__ __align__(128) u16 Vs[2][64][32];     // [d][key]   4 KB x2
    __shared__ __align__(128) u16 pls[8][16 * 32];   // per-wave P relayout tile

    const int lane = threadIdx.x & 31;
    const int wv   = threadIdx.x >> 5;
    const int hi   = lane >> 4;
    const int ln   = lane & 15;

    const int h  = blockIdx.y;
    const int b  = blockIdx.z;
    const int qb = blockIdx.x * 128;
    const int q0 = qb + wv * 16;
    const int qmax = q0 + 15;

    const size_t bh = (size_t)(b * 16 + h);
    const u16* qp = Q  + bh * 2048 * 64;
    const u16* kp = K  + bh * 2048 * 64;
    const u16* vp = Vt + bh * 64 * 2048;

    unsigned ksAddr[2], vsAddr[2];
#pragma unroll
    for (int p = 0; p < 2; ++p) {
        ksAddr[p] = (unsigned)(uintptr_t)(const void*)&Ks[p][0][0];
        vsAddr[p] = (unsigned)(uintptr_t)(const void*)&Vs[p][0][0];
    }

    v16bf qa[2];                                    // Q A-fragments (Dk=64 -> 2 k-steps)
#pragma unroll
    for (int t = 0; t < 2; ++t) {
        const u16* r = qp + (size_t)(q0 + ln) * 64 + t * 32 + hi * 8;
        qa[t] = ld_frag(r, r + 16);
    }

    v8f o[4];
#pragma unroll
    for (int jd = 0; jd < 4; ++jd) o[jd] = vzero();
    float mrow[8], lrow[8];
#pragma unroll
    for (int r = 0; r < 8; ++r) { mrow[r] = -1e30f; lrow[r] = 0.f; }

    const int nsteps = (qb + 128) >> 5;             // causal: block needs keys < qb+128

    if (threadIdx.x < 32) {                         // prologue: stage 0 K/V tiles
        tdm_load_2d(ksAddr[0], kp, 64, 32,   64, 2048u, 64);
        tdm_load_2d(vsAddr[0], vp, 32, 64, 2048u,   64, 2048);
    }

    for (int st = 0; st < nsteps; ++st) {
        const int k0 = st * 32;
        if (threadIdx.x < 32) {                     // overlap: DMA stage st+1
            if (st + 1 < nsteps) {
                const int k1 = k0 + 32;
                const int nb = (st + 1) & 1;
                tdm_load_2d(ksAddr[nb], kp + (size_t)k1 * 64, 64, 32,   64, 2048u, 64);
                tdm_load_2d(vsAddr[nb], vp + k1,              32, 64, 2048u,   64, 2048);
                __builtin_amdgcn_s_wait_tensorcnt(2);
            } else {
                __builtin_amdgcn_s_wait_tensorcnt(0);
            }
        }
        __syncthreads();                            // stage st tiles ready

        const int cb = st & 1;
        if (k0 <= qmax) {                           // wave has unmasked keys here
            // scores: two 16-col key tiles, two contraction steps each
            v8f s0 = vzero(), s1 = vzero();
#pragma unroll
            for (int t = 0; t < 2; ++t) {
                const u16* rk0 = &Ks[cb][ln][t * 32 + hi * 8];
                s0 = wmma_bf(qa[t], ld_frag(rk0, rk0 + 16), s0);
                const u16* rk1 = &Ks[cb][16 + ln][t * 32 + hi * 8];
                s1 = wmma_bf(qa[t], ld_frag(rk1, rk1 + 16), s1);
            }

            const int n0c = k0 + ln, n1c = k0 + 16 + ln;
#pragma unroll
            for (int r = 0; r < 8; ++r) {
                int qrow = q0 + hi * 8 + r;
                float a0 = s0[r] * 0.125f; if (n0c > qrow) a0 = -1e9f;  // 1/sqrt(64)
                float a1 = s1[r] * 0.125f; if (n1c > qrow) a1 = -1e9f;
                float rm = fmaxf(a0, a1);            // row max across 16 lanes
                rm = fmaxf(rm, __shfl_xor(rm, 1, 32));
                rm = fmaxf(rm, __shfl_xor(rm, 2, 32));
                rm = fmaxf(rm, __shfl_xor(rm, 4, 32));
                rm = fmaxf(rm, __shfl_xor(rm, 8, 32));
                float mnew = fmaxf(mrow[r], rm);
                float corr = __expf(mrow[r] - mnew);
                float p0 = __expf(a0 - mnew);
                float p1 = __expf(a1 - mnew);
                float rs = p0 + p1;                  // row sum across 16 lanes
                rs += __shfl_xor(rs, 1, 32);
                rs += __shfl_xor(rs, 2, 32);
                rs += __shfl_xor(rs, 4, 32);
                rs += __shfl_xor(rs, 8, 32);
                lrow[r] = lrow[r] * corr + rs;
                mrow[r] = mnew;
#pragma unroll
                for (int jd = 0; jd < 4; ++jd) o[jd][r] *= corr;
                pls[wv][(hi * 8 + r) * 32 + ln]      = f2bf(p0);  // P: C->A relayout
                pls[wv][(hi * 8 + r) * 32 + 16 + ln] = f2bf(p1);
            }

            // P as A-fragment (16x32), V^T as B-fragments: O += P*V
            const u16* pr = &pls[wv][ln * 32 + hi * 8];
            v16bf pa = ld_frag(pr, pr + 16);
#pragma unroll
            for (int jd = 0; jd < 4; ++jd) {
                const u16* rv = &Vs[cb][jd * 16 + ln][hi * 8];
                o[jd] = wmma_bf(pa, ld_frag(rv, rv + 16), o[jd]);
            }
        }
        __syncthreads();                            // buffer reusable for st+2 DMA
    }

#pragma unroll
    for (int r = 0; r < 8; ++r) {
        float inv = 1.0f / lrow[r];
        int s = q0 + hi * 8 + r;
        size_t row = ((size_t)b * 2048 + s) * 1024;
#pragma unroll
        for (int jd = 0; jd < 4; ++jd)
            O[row + h * 64 + jd * 16 + ln] = f2bf(o[jd][r] * inv);
    }
}

// ---------------------------------------------------------------------------
extern "C" void kernel_launch(void* const* d_in, const int* in_sizes, int n_in,
                              void* d_out, int out_size, void* d_ws, size_t ws_size,
                              hipStream_t stream)
{
    (void)in_sizes; (void)n_in; (void)out_size; (void)ws_size;
    const float* x  = (const float*)d_in[0];
    // d_in[1] = causal mask (tril), applied analytically in attn_fwd
    const float* wq = (const float*)d_in[2];
    const float* bq = (const float*)d_in[3];
    const float* wk = (const float*)d_in[4];
    const float* bk = (const float*)d_in[5];
    const float* wvp= (const float*)d_in[6];
    const float* bv = (const float*)d_in[7];
    const float* wo = (const float*)d_in[8];
    const float* bo = (const float*)d_in[9];
    float* out = (float*)d_out;

    char* ws = (char*)d_ws;
    u16* xbf  = (u16*)ws; ws += (size_t)4096 * 1024 * 2;
    u16* wqkv = (u16*)ws; ws += (size_t)3072 * 1024 * 2;
    u16* wob  = (u16*)ws; ws += (size_t)1024 * 1024 * 2;
    u16* Qb   = (u16*)ws; ws += (size_t)4096 * 1024 * 2;
    u16* Kb   = (u16*)ws; ws += (size_t)4096 * 1024 * 2;
    u16* Vtb  = (u16*)ws; ws += (size_t)4096 * 1024 * 2;
    u16* Ob   = (u16*)ws; ws += (size_t)4096 * 1024 * 2;

    auto cvt = [&](const float* s, u16* d, int n) {
        int n4 = n >> 2;
        cvt_bf16_kernel<<<(n4 + 255) / 256, 256, 0, stream>>>(s, d, n4);
    };
    cvt(x,   xbf,  4096 * 1024);
    cvt(wq,  wqkv,                        1024 * 1024);
    cvt(wk,  wqkv + (size_t)1024 * 1024,  1024 * 1024);
    cvt(wvp, wqkv + (size_t)2048 * 1024,  1024 * 1024);
    cvt(wo,  wob,  1024 * 1024);

    // Fused QKV projection: M=4096, N=3072, K=1024
    gemm_bf16<0><<<dim3(12, 64), 256, 0, stream>>>(
        xbf, wqkv, bq, bk, bv, Qb, Kb, Vtb, nullptr, 3072);

    // Causal flash attention: grid = (S/128, H, B)
    attn_fwd<<<dim3(16, 16, 2), 256, 0, stream>>>(Qb, Kb, Vtb, Ob);

    // Output projection: M=4096, N=1024, K=1024 -> fp32 d_out
    gemm_bf16<1><<<dim3(4, 64), 256, 0, stream>>>(
        Ob, wob, bo, nullptr, nullptr, nullptr, nullptr, nullptr, out, 1024);
}
